// DiaynDiscriminator_2903397892905
// MI455X (gfx1250) — matrix operand order repaired and checked by reading, not compiled, not run on hardware
//
#include <hip/hip_runtime.h>
#include <stdint.h>

#define BB      16384
#define OBS     512
#define GENC    216
#define HH      1024
#define CC      64
#define NF      8
#define IN_DIM  1240
#define IN_PAD  1248   // 39 * 32
#define K1_STEPS 39
#define K2_STEPS 32    // 1024 / 32

typedef __attribute__((ext_vector_type(16))) __bf16 v16bf;
typedef __attribute__((ext_vector_type(8)))  float  v8f;

union Frag {
  v16bf          bf;
  unsigned short us[16];
  unsigned int   u[8];
};

__device__ __forceinline__ unsigned short to_bf16(float f) {
  __bf16 h = (__bf16)f;                       // native v_cvt, RNE
  return __builtin_bit_cast(unsigned short, h);
}

// ---------------------------------------------------------------- factor select
__global__ void k_zero_counts(int* __restrict__ counts) {
  if (threadIdx.x < NF) counts[threadIdx.x] = 0;
}

__global__ void k_select(const float* __restrict__ graph,
                         int* __restrict__ fidx, int* __restrict__ counts) {
  int b = blockIdx.x * blockDim.x + threadIdx.x;
  if (b >= BB) return;
  int f = -1;
#pragma unroll
  for (int i = 0; i < NF; ++i)
    if (graph[(long)b * GENC + i] == 1.0f) f = i;   // last active factor wins
  fidx[b] = f;
  if (f >= 0) atomicAdd(&counts[f], 1);
}

__global__ void k_scan(const int* __restrict__ counts,
                       int* __restrict__ offsets, int* __restrict__ cursor) {
  if (threadIdx.x == 0) {
    int off = 0;
    for (int f = 0; f < NF; ++f) { offsets[f] = off; cursor[f] = off; off += counts[f]; }
  }
}

__global__ void k_scatter(const int* __restrict__ fidx, int* __restrict__ cursor,
                          int* __restrict__ rowlist, float* __restrict__ out) {
  int b = blockIdx.x * blockDim.x + threadIdx.x;
  if (b >= BB) return;
  int f = fidx[b];
  if (f >= 0) {
    int slot = atomicAdd(&cursor[f], 1);
    rowlist[slot] = b;
  } else {
    float4 z = make_float4(0.f, 0.f, 0.f, 0.f);
    float4* o = (float4*)(out + (long)b * CC);
#pragma unroll
    for (int c = 0; c < CC / 4; ++c) o[c] = z;
  }
}

// ---------------------------------------------------------------- weight pack
// dst[((f*ksteps + ks)*N + col)*32 + kk] = bf16(W[f][ks*32+kk][col]), zero-padded.
// A lane's B-fragment (16 bf16, K = e + 16*half) is then 32 contiguous bytes.
__global__ void __launch_bounds__(256)
k_pack(const float* __restrict__ W, unsigned short* __restrict__ dst,
       int K, int ksteps, int N) {
  const int f  = blockIdx.z;
  const int ks = blockIdx.y;
  const int colBase = blockIdx.x * 64;
  __shared__ float tile[32][65];               // pad 65 -> conflict-free transpose
  const int tid = threadIdx.x;
  const float* Wf = W + (size_t)f * K * N;
  for (int i = tid; i < 32 * 64; i += 256) {   // coalesced reads (col fastest)
    int kk = i >> 6;
    int c  = i & 63;
    int k  = ks * 32 + kk;
    tile[kk][c] = (k < K) ? Wf[(size_t)k * N + colBase + c] : 0.f;
  }
  __syncthreads();
  unsigned short* out = dst + (((size_t)(f * ksteps + ks)) * N + colBase) * 32;
  for (int i = tid; i < 32 * 64; i += 256) {   // coalesced writes (kk fastest)
    int c  = i >> 5;
    int kk = i & 31;
    out[(size_t)c * 32 + kk] = to_bf16(tile[kk][c]);
  }
}

__device__ __forceinline__ void load_bfrag_packed(Frag& b, const unsigned short* p) {
  const uint4* q = (const uint4*)p;            // 2 x global_load_b128
  uint4 a0 = q[0];
  uint4 a1 = q[1];
  b.u[0] = a0.x; b.u[1] = a0.y; b.u[2] = a0.z; b.u[3] = a0.w;
  b.u[4] = a1.x; b.u[5] = a1.y; b.u[6] = a1.z; b.u[7] = a1.w;
}

// ---------------------------------------------------------------- fused MLP tile
template <bool PACKED>
__global__ void __launch_bounds__(256)
k_mlp(const float* __restrict__ graph, const float* __restrict__ state,
      const float* __restrict__ nstate,
      const float* __restrict__ W1, const float* __restrict__ b1,
      const float* __restrict__ W2, const float* __restrict__ b2,
      const float* __restrict__ W3, const float* __restrict__ b3,
      const unsigned short* __restrict__ pW1,
      const unsigned short* __restrict__ pW2,
      const unsigned short* __restrict__ pW3,
      const int* __restrict__ counts, const int* __restrict__ offsets,
      const int* __restrict__ rowlist, float* __restrict__ out) {
  const int f = blockIdx.y;
  const int t = blockIdx.x;
  const int cnt = counts[f];
  if (t * 16 >= cnt) return;
  const int off    = offsets[f];
  const int nvalid = min(16, cnt - t * 16);

  extern __shared__ char smem_raw[];
  unsigned short* xA = (unsigned short*)smem_raw;   // 16 x IN_PAD bf16 (39936 B)
  unsigned short* h1 = xA + 16 * IN_PAD;            // 16 x HH bf16     (32768 B)
  int* rows_s = (int*)(h1 + 16 * HH);               // 16 row indices

  const int tid = threadIdx.x;
  if (tid < 16) {
    int slot = t * 16 + tid;
    rows_s[tid] = rowlist[off + (slot < cnt ? slot : 0)];  // pad with slot 0
  }
  __syncthreads();

  // gather x = [graph | state | next_state] for the 16 rows, bf16, zero-padded K
  for (int r = 0; r < 16; ++r) {
    const int row = rows_s[r];
    for (int k = tid; k < IN_PAD; k += 256) {
      float v;
      if (k < GENC)            v = graph [(long)row * GENC + k];
      else if (k < GENC + OBS) v = state [(long)row * OBS  + (k - GENC)];
      else if (k < IN_DIM)     v = nstate[(long)row * OBS  + (k - GENC - OBS)];
      else                     v = 0.0f;
      xA[r * IN_PAD + k] = to_bf16(v);
    }
  }
  __syncthreads();

  const int wave = tid >> 5;
  const int lane = tid & 31;
  const int half = lane >> 4;   // lane half selects K/M group per ISA layouts
  const int nIdx = lane & 15;   // N (B/D cols) and M (A rows) position
  const int colBase = wave * 128;
  const v8f vzero = {0.f, 0.f, 0.f, 0.f, 0.f, 0.f, 0.f, 0.f};

  v8f acc[8];

  // ---------------- layer 1: h1 = relu(x @ W1[f] + b1[f]), K -> 39x32
#pragma unroll
  for (int nt = 0; nt < 8; ++nt) acc[nt] = vzero;
  {
    const float* W1f = W1 + (long)f * IN_DIM * HH;
    for (int ks = 0; ks < K1_STEPS; ++ks) {
      const int kbase = ks * 32;
      Frag a;
#pragma unroll
      for (int p = 0; p < 8; ++p) {       // A k order used here (contiguous): pair p -> K = 2p + 16*half
        int K = kbase + 2 * p + 16 * half;
        a.u[p] = *(const unsigned int*)(xA + nIdx * IN_PAD + K);
      }
#pragma unroll
      for (int nt = 0; nt < 8; ++nt) {
        const int col = colBase + nt * 16 + nIdx;
        Frag bfr;
        if (PACKED) {
          load_bfrag_packed(bfr,
              pW1 + (((size_t)(f * K1_STEPS + ks)) * HH + col) * 32 + 16 * half);
        } else {
#pragma unroll
          for (int e = 0; e < 16; ++e) {  // B k order used here (matches A): lanes 0-15 K=e, lanes 16-31 K=e+16
            int k = kbase + e + 16 * half;
            k = (k < IN_DIM) ? k : (IN_DIM - 1);   // clamp; x pad is zero anyway
            bfr.us[e] = to_bf16(W1f[(long)k * HH + col]);
          }
        }
        acc[nt] = __builtin_amdgcn_wmma_f32_16x16x32_bf16(
            false, a.bf, false, bfr.bf, (short)0, acc[nt], false, false);
      }
    }
#pragma unroll
    for (int nt = 0; nt < 8; ++nt) {
      const int col = colBase + nt * 16 + nIdx;
      const float bias = b1[f * HH + col];
#pragma unroll
      for (int v = 0; v < 8; ++v) {       // D layout: VGPR v -> M = v + 8*half
        int m = v + 8 * half;
        float val = acc[nt][v] + bias;
        val = val > 0.f ? val : 0.f;
        h1[m * HH + col] = to_bf16(val);
      }
    }
  }
  __syncthreads();

  // ---------------- layer 2: h2 = relu(h1 @ W2[f] + b2[f]), K = 1024 -> 32x32
  unsigned short* h2 = xA;   // reuse x buffer (no longer read)
#pragma unroll
  for (int nt = 0; nt < 8; ++nt) acc[nt] = vzero;
  {
    const float* W2f = W2 + (long)f * HH * HH;
    for (int ks = 0; ks < K2_STEPS; ++ks) {
      const int kbase = ks * 32;
      Frag a;
#pragma unroll
      for (int p = 0; p < 8; ++p) {
        int K = kbase + 2 * p + 16 * half;
        a.u[p] = *(const unsigned int*)(h1 + nIdx * HH + K);
      }
#pragma unroll
      for (int nt = 0; nt < 8; ++nt) {
        const int col = colBase + nt * 16 + nIdx;
        Frag bfr;
        if (PACKED) {
          load_bfrag_packed(bfr,
              pW2 + (((size_t)(f * K2_STEPS + ks)) * HH + col) * 32 + 16 * half);
        } else {
#pragma unroll
          for (int e = 0; e < 16; ++e) {
            int k = kbase + e + 16 * half;
            bfr.us[e] = to_bf16(W2f[(long)k * HH + col]);
          }
        }
        acc[nt] = __builtin_amdgcn_wmma_f32_16x16x32_bf16(
            false, a.bf, false, bfr.bf, (short)0, acc[nt], false, false);
      }
    }
#pragma unroll
    for (int nt = 0; nt < 8; ++nt) {
      const int col = colBase + nt * 16 + nIdx;
      const float bias = b2[f * HH + col];
#pragma unroll
      for (int v = 0; v < 8; ++v) {
        int m = v + 8 * half;
        float val = acc[nt][v] + bias;
        val = val > 0.f ? val : 0.f;
        h2[m * HH + col] = to_bf16(val);
      }
    }
  }
  __syncthreads();

  // ---------------- layer 3: out = h2 @ W3[f] + b3[f], N = 64 -> waves 0..3
  if (wave < 4) {
    v8f acc3 = vzero;
    const int col = wave * 16 + nIdx;
    const float* W3f = W3 + (long)f * HH * CC;
    for (int ks = 0; ks < K2_STEPS; ++ks) {
      const int kbase = ks * 32;
      Frag a;
#pragma unroll
      for (int p = 0; p < 8; ++p) {
        int K = kbase + 2 * p + 16 * half;
        a.u[p] = *(const unsigned int*)(h2 + nIdx * HH + K);
      }
      Frag bfr;
      if (PACKED) {
        load_bfrag_packed(bfr,
            pW3 + (((size_t)(f * K2_STEPS + ks)) * CC + col) * 32 + 16 * half);
      } else {
#pragma unroll
        for (int e = 0; e < 16; ++e) {
          int k = kbase + e + 16 * half;
          bfr.us[e] = to_bf16(W3f[(long)k * CC + col]);
        }
      }
      acc3 = __builtin_amdgcn_wmma_f32_16x16x32_bf16(
          false, a.bf, false, bfr.bf, (short)0, acc3, false, false);
    }
    const float bias = b3[f * CC + col];
#pragma unroll
    for (int v = 0; v < 8; ++v) {
      int m = v + 8 * half;
      if (m < nvalid) out[(long)rows_s[m] * CC + col] = acc3[v] + bias;
    }
  }
}

// ---------------------------------------------------------------- launch
extern "C" void kernel_launch(void* const* d_in, const int* in_sizes, int n_in,
                              void* d_out, int out_size, void* d_ws, size_t ws_size,
                              hipStream_t stream) {
  (void)in_sizes; (void)n_in; (void)out_size;
  const float* graph  = (const float*)d_in[0];
  const float* state  = (const float*)d_in[1];
  const float* nstate = (const float*)d_in[2];
  const float* W1 = (const float*)d_in[3];
  const float* b1 = (const float*)d_in[4];
  const float* W2 = (const float*)d_in[5];
  const float* b2 = (const float*)d_in[6];
  const float* W3 = (const float*)d_in[7];
  const float* b3 = (const float*)d_in[8];
  float* out = (float*)d_out;

  char* base = (char*)d_ws;
  int* counts  = (int*)base;      // 8
  int* offsets = counts + NF;     // 8
  int* cursor  = offsets + NF;    // 8
  int* fidx    = cursor + NF;     // B
  int* rowlist = fidx + BB;       // B

  const size_t intBytes = ((size_t)(3 * NF + 2 * BB) * sizeof(int) + 255) & ~(size_t)255;
  const size_t w1e = (size_t)NF * K1_STEPS * HH * 32;   // bf16 elements
  const size_t w2e = (size_t)NF * K2_STEPS * HH * 32;
  const size_t w3e = (size_t)NF * K2_STEPS * CC * 32;
  unsigned short* pW1 = (unsigned short*)(base + intBytes);
  unsigned short* pW2 = pW1 + w1e;
  unsigned short* pW3 = pW2 + w2e;
  const bool packed = ws_size >= intBytes + (w1e + w2e + w3e) * sizeof(unsigned short);

  k_zero_counts<<<1, 32, 0, stream>>>(counts);
  k_select<<<BB / 256, 256, 0, stream>>>(graph, fidx, counts);
  k_scan<<<1, 32, 0, stream>>>(counts, offsets, cursor);
  k_scatter<<<BB / 256, 256, 0, stream>>>(fidx, cursor, rowlist, out);

  size_t smem = (size_t)(16 * IN_PAD + 16 * HH) * sizeof(unsigned short)
              + 16 * sizeof(int);
  dim3 grid(BB / 16, NF);

  if (packed) {
    k_pack<<<dim3(HH / 64, K1_STEPS, NF), 256, 0, stream>>>(W1, pW1, IN_DIM, K1_STEPS, HH);
    k_pack<<<dim3(HH / 64, K2_STEPS, NF), 256, 0, stream>>>(W2, pW2, HH, K2_STEPS, HH);
    k_pack<<<dim3(CC / 64, K2_STEPS, NF), 256, 0, stream>>>(W3, pW3, HH, K2_STEPS, CC);
    k_mlp<true><<<grid, 256, smem, stream>>>(graph, state, nstate,
                                             W1, b1, W2, b2, W3, b3,
                                             pW1, pW2, pW3,
                                             counts, offsets, rowlist, out);
  } else {
    k_mlp<false><<<grid, 256, smem, stream>>>(graph, state, nstate,
                                              W1, b1, W2, b2, W3, b3,
                                              nullptr, nullptr, nullptr,
                                              counts, offsets, rowlist, out);
  }
}
